// encoding2_3968549781738
// MI455X (gfx1250) — matrix operand order repaired and checked by reading, not compiled, not run on hardware
//
#include <hip/hip_runtime.h>
#include <hip/hip_bf16.h>

#define NNODES 50000
#define HID    256
#define INDIM  500
#define KPAD   512

typedef __attribute__((ext_vector_type(16))) __bf16         v16bf;
typedef __attribute__((ext_vector_type(8)))  float          v8f;
typedef __attribute__((ext_vector_type(16))) unsigned short v16u;
typedef __attribute__((ext_vector_type(8)))  unsigned short v8u;

__device__ __forceinline__ unsigned short f2bf(float f) {
  unsigned int u = __float_as_uint(f);
  if ((u & 0x7f800000u) == 0x7f800000u) return (unsigned short)(u >> 16); // inf/nan
  u += 0x7fffu + ((u >> 16) & 1u);                                        // RNE
  return (unsigned short)(u >> 16);
}

__device__ __forceinline__ v8f wmma_bf16(v16bf a, v16bf b, v8f c) {
  return __builtin_amdgcn_wmma_f32_16x16x32_bf16(false, a, false, b,
                                                 (short)0, c, false, false);
}

// ---- f32 [rows,kin] -> bf16 [rows,kout], zero-padded in K -------------------
__global__ void __launch_bounds__(256)
cvt_bf16_pad(const float* __restrict__ in, unsigned short* __restrict__ out,
             int rows, int kin, int kout) {
  long long i = (long long)blockIdx.x * blockDim.x + threadIdx.x;
  long long tot = (long long)rows * kout;
  if (i >= tot) return;
  int r = (int)(i / kout), k = (int)(i % kout);
  out[i] = (k < kin) ? f2bf(in[(long long)r * kin + k]) : (unsigned short)0;
}

__global__ void __launch_bounds__(256)
fill_zero(float* __restrict__ p, long long n) {
  long long i = (long long)blockIdx.x * blockDim.x + threadIdx.x;
  if (i < n) p[i] = 0.0f;
}

// ---- WMMA GEMM, register-blocked 16x64 macro-tile per wave ------------------
// C[M,Nout] = A[M,K](bf16) x W[Nout,K](bf16)^T (+bias)(relu)
// grid = (M/16, Nout/64), block = 32. Per K-step: 1 A-frag (2x b128) shared by
// 4 B-frags (8x b128) -> 4 WMMAs; 2.5 VMEM/WMMA instead of 4.
// A-frag layout (16x32 bf16, contiguous K): lane = M%16 (+hsel), halves[0..7] =
// K kb..kb+7, halves[8..15] = K kb+8..kb+15, kb=(lane>>4)*16. B[k][n]=W[n][k]:
// lane holds column n, 16 contiguous bf16 at kb=(lane>>4)*16.
__global__ void __launch_bounds__(32)
gemm_bf16_wmma(const unsigned short* __restrict__ A,
               const unsigned short* __restrict__ W,
               const float* __restrict__ bias, int has_bias, int relu,
               float* __restrict__ C, int K, int Nout) {
  const int lane  = threadIdx.x;
  const int mbase = blockIdx.x * 16;
  const int nbase = blockIdx.y * 64;
  const int ml    = lane & 15;
  const int hsel  = lane >> 4;
  const unsigned short* arow = A + (size_t)(mbase + ml) * K + (hsel << 4);
  const unsigned short* wr0  = W + (size_t)(nbase +  0 + ml) * K + (hsel << 4);
  const unsigned short* wr1  = W + (size_t)(nbase + 16 + ml) * K + (hsel << 4);
  const unsigned short* wr2  = W + (size_t)(nbase + 32 + ml) * K + (hsel << 4);
  const unsigned short* wr3  = W + (size_t)(nbase + 48 + ml) * K + (hsel << 4);
  v8f acc0 = {}, acc1 = {}, acc2 = {}, acc3 = {};
  for (int k = 0; k < K; k += 32) {
    v8u alo = *(const v8u*)(arow + k);        // K = kb .. kb+7
    v8u ahi = *(const v8u*)(arow + k + 8);    // K = kb+8 .. kb+15
    v16u au;
#pragma unroll
    for (int i = 0; i < 8; ++i) { au[i] = alo[i]; au[i + 8] = ahi[i]; }
    v16bf a  = __builtin_bit_cast(v16bf, au);
    v16bf b0 = __builtin_bit_cast(v16bf, *(const v16u*)(wr0 + k));
    v16bf b1 = __builtin_bit_cast(v16bf, *(const v16u*)(wr1 + k));
    v16bf b2 = __builtin_bit_cast(v16bf, *(const v16u*)(wr2 + k));
    v16bf b3 = __builtin_bit_cast(v16bf, *(const v16u*)(wr3 + k));
    acc0 = wmma_bf16(a, b0, acc0);
    acc1 = wmma_bf16(a, b1, acc1);
    acc2 = wmma_bf16(a, b2, acc2);
    acc3 = wmma_bf16(a, b3, acc3);
  }
  v8f accs[4] = {acc0, acc1, acc2, acc3};
#pragma unroll
  for (int jt = 0; jt < 4; ++jt) {
    const int col = nbase + jt * 16 + ml;
    const float bv = has_bias ? bias[col] : 0.0f;
#pragma unroll
    for (int j = 0; j < 8; ++j) {
      int row = mbase + j + (hsel << 3);      // C/D layout: vgpr j -> M=j(+8)
      float v = accs[jt][j] + bv;
      if (relu) v = fmaxf(v, 0.0f);
      C[(size_t)row * Nout + col] = v;
    }
  }
}

// ---- Gram: G[HID,HID] += C^T C, split-K, j-blocked x4, f32 atomics ----------
// grid = (HID/16, HID/64, Z). A[m][k]=C[nb+k][ib+m] gathered once per K-step,
// reused across 4 j-tiles; B[k][n]=C[nb+k][jb+n].
__global__ void __launch_bounds__(32)
gram_bf16_wmma(const unsigned short* __restrict__ Cbf, float* __restrict__ G,
               int Ntot) {
  const int lane = threadIdx.x;
  const int ib = blockIdx.x * 16, jb = blockIdx.y * 64;
  const int ml = lane & 15, hsel = lane >> 4;
  const int ktiles = (Ntot + 31) / 32;
  const int per = (ktiles + gridDim.z - 1) / gridDim.z;
  int kt0 = blockIdx.z * per;
  int kt1 = kt0 + per; if (kt1 > ktiles) kt1 = ktiles;
  v8f acc0 = {}, acc1 = {}, acc2 = {}, acc3 = {};
  for (int kt = kt0; kt < kt1; ++kt) {
    const int nb = kt * 32;
    v16u au;
#pragma unroll
    for (int t = 0; t < 8; ++t) {
      int r0 = nb + hsel * 8 + t;
      int r1 = nb + 16 + hsel * 8 + t;
      au[t]     = (r0 < Ntot) ? Cbf[(size_t)r0 * HID + ib + ml] : (unsigned short)0;
      au[t + 8] = (r1 < Ntot) ? Cbf[(size_t)r1 * HID + ib + ml] : (unsigned short)0;
    }
    v16bf a = __builtin_bit_cast(v16bf, au);
    v8f* accp[4] = {&acc0, &acc1, &acc2, &acc3};
#pragma unroll
    for (int jt = 0; jt < 4; ++jt) {
      v16u bu;
#pragma unroll
      for (int t = 0; t < 16; ++t) {
        int r = nb + hsel * 8 + (t < 8 ? t : t + 8);
        bu[t] = (r < Ntot) ? Cbf[(size_t)r * HID + jb + jt * 16 + ml]
                           : (unsigned short)0;
      }
      v16bf b = __builtin_bit_cast(v16bf, bu);
      *accp[jt] = wmma_bf16(a, b, *accp[jt]);
    }
  }
  v8f accs[4] = {acc0, acc1, acc2, acc3};
#pragma unroll
  for (int jt = 0; jt < 4; ++jt) {
#pragma unroll
    for (int j = 0; j < 8; ++j) {
      int i = ib + j + (hsel << 3);
      __hip_atomic_fetch_add(&G[i * HID + jb + jt * 16 + ml], accs[jt][j],
                             __ATOMIC_RELAXED, __HIP_MEMORY_SCOPE_AGENT);
    }
  }
}

// ---- SpMM segment-sum: S[dst] += w * H[src]; 64 threads/edge, float4 --------
__global__ void __launch_bounds__(256)
spmm_atomic(const int* __restrict__ ei, const float* __restrict__ ew,
            const float* __restrict__ Hsrc, float* __restrict__ S, int E) {
  int e = blockIdx.x * 4 + (threadIdx.x >> 6);
  if (e >= E) return;
  int t = threadIdx.x & 63;
  int dst = ei[e];        // edge_index[0]
  int src = ei[E + e];    // edge_index[1]
  float w = ew[e];
  __builtin_prefetch(ei + e + 1024, 0, 1);      // global_prefetch_b8 (edge stream)
  __builtin_prefetch(ei + E + e + 1024, 0, 1);
  float4 v = ((const float4*)(Hsrc + (size_t)src * HID))[t];  // L2-resident gather
  float* o = S + (size_t)dst * HID + t * 4;
  __hip_atomic_fetch_add(o + 0, w * v.x, __ATOMIC_RELAXED, __HIP_MEMORY_SCOPE_AGENT);
  __hip_atomic_fetch_add(o + 1, w * v.y, __ATOMIC_RELAXED, __HIP_MEMORY_SCOPE_AGENT);
  __hip_atomic_fetch_add(o + 2, w * v.z, __ATOMIC_RELAXED, __HIP_MEMORY_SCOPE_AGENT);
  __hip_atomic_fetch_add(o + 3, w * v.w, __ATOMIC_RELAXED, __HIP_MEMORY_SCOPE_AGENT);
}

// ---- out = a*S + b*H0 (float4) ----------------------------------------------
__global__ void __launch_bounds__(256)
axpby4(float* __restrict__ out, const float* __restrict__ S,
       const float* __restrict__ H0, float a, float b, long long n4) {
  long long i = (long long)blockIdx.x * blockDim.x + threadIdx.x;
  if (i >= n4) return;
  float4 s = ((const float4*)S)[i], h = ((const float4*)H0)[i];
  float4 r = { a * s.x + b * h.x, a * s.y + b * h.y,
               a * s.z + b * h.z, a * s.w + b * h.w };
  ((float4*)out)[i] = r;
}

// ---- de_propagate combine: out = cz*C + cs*S + ct*T + c0*H0 -----------------
__global__ void __launch_bounds__(256)
de_combine4(float* __restrict__ out, const float* __restrict__ Cc,
            const float* __restrict__ S, const float* __restrict__ T,
            const float* __restrict__ H0, float cz, float cs, float ct,
            float c0, long long n4) {
  long long i = (long long)blockIdx.x * blockDim.x + threadIdx.x;
  if (i >= n4) return;
  float4 c = ((const float4*)Cc)[i], s = ((const float4*)S)[i];
  float4 t = ((const float4*)T)[i],  h = ((const float4*)H0)[i];
  float4 r = { cz * c.x + cs * s.x + ct * t.x + c0 * h.x,
               cz * c.y + cs * s.y + ct * t.y + c0 * h.y,
               cz * c.z + cs * s.z + ct * t.z + c0 * h.z,
               cz * c.w + cs * s.w + ct * t.w + c0 * h.w };
  ((float4*)out)[i] = r;
}

// ---- row L2 normalize (in-place), one block per row -------------------------
__global__ void __launch_bounds__(HID)
rownorm(float* __restrict__ out) {
  __shared__ float red[HID];
  const int row = blockIdx.x, t = threadIdx.x;
  float v = out[(size_t)row * HID + t];
  red[t] = v * v;
  __syncthreads();
  for (int s = HID / 2; s > 0; s >>= 1) {
    if (t < s) red[t] += red[t + s];
    __syncthreads();
  }
  float inv = 1.0f / fmaxf(sqrtf(red[0]), 1e-12f);
  out[(size_t)row * HID + t] = v * inv;
}

extern "C" void kernel_launch(void* const* d_in, const int* in_sizes, int n_in,
                              void* d_out, int out_size, void* d_ws, size_t ws_size,
                              hipStream_t stream) {
  (void)n_in; (void)out_size; (void)ws_size;
  const float* X  = (const float*)d_in[0];
  const int*   EI = (const int*)d_in[1];
  const float* EW = (const float*)d_in[2];
  const float* W1 = (const float*)d_in[3];
  const float* b1 = (const float*)d_in[4];
  const float* W2 = (const float*)d_in[5];
  const float* b2 = (const float*)d_in[6];
  const float* Wf = (const float*)d_in[7];
  const float* bf = (const float*)d_in[8];
  const int N = NNODES;
  const int E = in_sizes[2];

  // ---- workspace carve (all 256B aligned) ----
  char* w = (char*)d_ws;
  auto take = [&](size_t bytes) -> char* {
    char* p = w; w += (bytes + 255) & ~(size_t)255; return p;
  };
  float*          H0  = (float*)take((size_t)N * HID * 4);
  float*          H   = (float*)take((size_t)N * HID * 4);
  float*          S   = (float*)take((size_t)N * HID * 4);
  unsigned short* Abf = (unsigned short*)take((size_t)N * HID * 2);
  unsigned short* W1b = (unsigned short*)take((size_t)HID * KPAD * 2);
  unsigned short* W2b = (unsigned short*)take((size_t)HID * HID * 2);
  unsigned short* Wfb = (unsigned short*)take((size_t)HID * HID * 2);
  unsigned short* Gb  = (unsigned short*)take((size_t)HID * HID * 2);
  float*          Gf  = (float*)take((size_t)HID * HID * 4);
  unsigned short* Xbf = (unsigned short*)S;  // alias: N*KPAD*2 == N*HID*4 bytes
  float*          T   = (float*)d_out;       // reuse output as f32 GEMM scratch

  const long long n4  = (long long)N * HID / 4;
  const dim3 b256(256), b32(32);
  const dim3 gemm_grid((N + 15) / 16, HID / 64);   // 16x64 macro-tiles
  const dim3 gram_grid(HID / 16, HID / 64, 98);    // 98*16 >= ceil(N/32)=1563
  const unsigned ew_grid   = (unsigned)(((long long)N * HID / 4 + 255) / 256);
  const unsigned spmm_grid = (unsigned)((E + 3) / 4);

  auto cvt = [&](const float* in, unsigned short* o, int rows, int kin, int kout) {
    long long tot = (long long)rows * kout;
    cvt_bf16_pad<<<dim3((unsigned)((tot + 255) / 256)), b256, 0, stream>>>(in, o, rows, kin, kout);
  };
  auto zero = [&](float* p, long long n) {
    fill_zero<<<dim3((unsigned)((n + 255) / 256)), b256, 0, stream>>>(p, n);
  };
  auto spmm = [&](const float* Hin) {
    zero(S, (long long)N * HID);
    spmm_atomic<<<dim3(spmm_grid), b256, 0, stream>>>(EI, EW, Hin, S, E);
  };

  // ---- first_layer: H0 = relu(X@W1^T + b1) @ W2^T + b2 ----
  cvt(X,  Xbf, N,   INDIM, KPAD);
  cvt(W1, W1b, HID, INDIM, KPAD);
  gemm_bf16_wmma<<<gemm_grid, b32, 0, stream>>>(Xbf, W1b, b1, 1, 1, T, KPAD, HID);
  cvt(T,  Abf, N,   HID, HID);
  cvt(W2, W2b, HID, HID, HID);
  gemm_bf16_wmma<<<gemm_grid, b32, 0, stream>>>(Abf, W2b, b2, 1, 0, H0, HID, HID);
  cvt(Wf, Wfb, HID, HID, HID);
  axpby4<<<dim3(ew_grid), b256, 0, stream>>>(H, H0, H0, 1.0f, 0.0f, n4);  // H = H0

  // ---- 3 propagation iterations ----
  for (int it = 0; it < 3; ++it) {
    // H = 0.5*spmm(H) + H0
    spmm(H);
    axpby4<<<dim3(ew_grid), b256, 0, stream>>>(H, S, H0, 0.5f, 1.0f, n4);
    // H = 0.5*spmm(H) + 0.5*H0
    spmm(H);
    axpby4<<<dim3(ew_grid), b256, 0, stream>>>(H, S, H0, 0.5f, 0.5f, n4);
    // de_propagate(C=H, C0=H0):
    spmm(H);                               // s-term
    cvt(H, Abf, N, HID, HID);
    zero(Gf, (long long)HID * HID);
    gram_bf16_wmma<<<gram_grid, b32, 0, stream>>>(Abf, Gf, N);   // G = C^T C
    cvt(Gf, Gb, HID, HID, HID);
    gemm_bf16_wmma<<<gemm_grid, b32, 0, stream>>>(Abf, Gb, nullptr, 0, 0,
                                                  T, HID, HID);  // T = C@G (G sym)
    // H = 0.8*C + 0.25*S - 0.05*T + 0.5*H0
    de_combine4<<<dim3(ew_grid), b256, 0, stream>>>(H, H, S, T, H0,
                                                    0.8f, 0.25f, -0.05f, 0.5f, n4);
  }

  // ---- final layer + row normalize ----
  cvt(H, Abf, N, HID, HID);
  gemm_bf16_wmma<<<gemm_grid, b32, 0, stream>>>(Abf, Wfb, bf, 1, 0, T, HID, HID);
  rownorm<<<dim3((unsigned)N), dim3(HID), 0, stream>>>(T);
}
